// QANet_19473381720744
// MI455X (gfx1250) — matrix operand order repaired and checked by reading, not compiled, not run on hardware
//
#include <hip/hip_runtime.h>
#include <hip/hip_bf16.h>

// ---------------------------------------------------------------------------
// QANet forward for MI455X (gfx1250, wave32, WMMA).
// All dense matmuls (pointwise convs, Wqs/Wo/fc, highway, CQ attention,
// attention scores & P@V) run on v_wmma_f32_16x16x32_f16 with f32 accum.
// Attention is fully fused (scores -> masked softmax -> P@V) in LDS per
// 64-query tile; the K/V head slice is staged in LDS so every WMMA fragment
// is built from aligned ds_load_b64/b128 vector reads.
// ---------------------------------------------------------------------------

#define BBATCH 32
#define LC 400
#define LQ 50
#define DD 128
#define NH 8
#define EE 364
#define EP 384
#define NEGV (-1e30f)

typedef __attribute__((ext_vector_type(16))) _Float16 v16h;
typedef __attribute__((ext_vector_type(8)))  _Float16 v8h;
typedef __attribute__((ext_vector_type(4)))  _Float16 v4h;
typedef __attribute__((ext_vector_type(8)))  float    v8f;

static inline dim3 g1d(long n) { return dim3((unsigned)((n + 255) / 256)); }

// ---------------------------------------------------------------------------
// Generic WMMA GEMM:  C[M,N] = act(A[M,K](f32) * B[KP,NP](f16) + bias) + Res
// A row-major (lda), B row-major padded (zeros beyond real K/N), C row-major.
// Block = 128 threads (4 waves), tile 64x64, K-step 32.
// ---------------------------------------------------------------------------
#define GBM 64
#define GBN 64
#define GBK 32
#define LDT 48  // LDS row stride in halves (96B, 16B-aligned fragments)

__global__ __launch_bounds__(128)
void gemm_f16_kernel(const float* __restrict__ A, long strideA, int lda,
                     const _Float16* __restrict__ Bm, long strideB,
                     float* __restrict__ C, long strideC, int ldc,
                     const float* __restrict__ bias, int biasN,
                     const float* __restrict__ Res, long strideR,
                     int M, int N, int K, int KP, int NP, int act) {
  __shared__ _Float16 sA[GBM * LDT];
  __shared__ _Float16 sB[GBN * LDT];
  const int tid = threadIdx.x, lane = tid & 31, wv = tid >> 5;
  const int bz = blockIdx.z;
  const int Mb = blockIdx.y * GBM, Nb = blockIdx.x * GBN;
  A += bz * strideA;
  Bm += bz * strideB;
  C += bz * strideC;
  if (Res) Res += bz * strideR;
  v8f acc[2][2] = {};
  const int ms = (wv >> 1) * 32, ns = (wv & 1) * 32;
  const int l15 = lane & 15;
  const int kbA = (lane >= 16) ? 16 : 0;  // A fragment k-split (contiguous k = 16h + s, same as B)
  const int kbB = (lane >= 16) ? 16 : 0;  // B fragment k-split
  const bool fullM = (Mb + GBM <= M);
  const bool fullN = (Nb + GBN <= NP);

  for (int kc = 0; kc < KP; kc += GBK) {
    // ---- stage A (f32 -> f16), vectorized fast path on interior tiles ----
    if (fullM && (kc + GBK <= K)) {
      for (int idx = tid; idx < GBM * 8; idx += 128) {
        int r = idx >> 3, kg = (idx & 7) << 2;
        const float4 v = *(const float4*)&A[(long)(Mb + r) * lda + kc + kg];
        v4h h;
        h[0] = (_Float16)v.x; h[1] = (_Float16)v.y;
        h[2] = (_Float16)v.z; h[3] = (_Float16)v.w;
        *(v4h*)&sA[r * LDT + kg] = h;
      }
    } else {
      for (int idx = tid; idx < GBM * GBK; idx += 128) {
        int r = idx >> 5, k = idx & 31;
        int m = Mb + r, kk = kc + k;
        float v = (m < M && kk < K) ? A[(long)m * lda + kk] : 0.f;
        sA[r * LDT + k] = (_Float16)v;
      }
    }
    // ---- stage B transposed: sB[n][k] = B[k][n] (KP is a multiple of 32) --
    if (fullN) {
      for (int idx = tid; idx < GBK * 8; idx += 128) {
        int k = idx >> 3, ng = (idx & 7) << 3;
        const v8h v = *(const v8h*)&Bm[(long)(kc + k) * NP + Nb + ng];
#pragma unroll
        for (int t = 0; t < 8; ++t) sB[(ng + t) * LDT + k] = v[t];
      }
    } else {
      for (int idx = tid; idx < GBN * GBK; idx += 128) {
        int n = idx & 63, k = idx >> 6;
        int nn = Nb + n;
        _Float16 v = (nn < NP) ? Bm[(long)(kc + k) * NP + nn] : (_Float16)0.f;
        sB[n * LDT + k] = v;
      }
    }
    // prefetch next A tile rows into cache while we compute
    if (kc + GBK < K) {
      int pm = Mb + (tid & 63);
      if (pm < M) __builtin_prefetch(&A[(long)pm * lda + kc + GBK], 0, 1);
    }
    __syncthreads();
#pragma unroll
    for (int i = 0; i < 2; ++i) {
      v16h a;
      {
        const v8h lo = *(const v8h*)&sA[(ms + i * 16 + l15) * LDT + kbA];
        const v8h hi = *(const v8h*)&sA[(ms + i * 16 + l15) * LDT + kbA + 8];
#pragma unroll
        for (int t = 0; t < 8; ++t) { a[t] = lo[t]; a[t + 8] = hi[t]; }
      }
#pragma unroll
      for (int j = 0; j < 2; ++j) {
        const v16h b = *(const v16h*)&sB[(ns + j * 16 + l15) * LDT + kbB];
        acc[i][j] = __builtin_amdgcn_wmma_f32_16x16x32_f16(
            false, a, false, b, (short)0, acc[i][j], false, false);
      }
    }
    __syncthreads();
  }
  const int mo = (lane >= 16) ? 8 : 0;
#pragma unroll
  for (int i = 0; i < 2; ++i)
#pragma unroll
    for (int j = 0; j < 2; ++j)
#pragma unroll
      for (int r = 0; r < 8; ++r) {
        int m = Mb + ms + i * 16 + r + mo;
        int n = Nb + ns + j * 16 + l15;
        if (m < M && n < N) {
          float v = acc[i][j][r];
          if (bias && n < biasN) v += bias[n];
          if (act == 1) v = fmaxf(v, 0.f);
          else if (act == 2) v = 1.f / (1.f + __expf(-v));
          if (Res) v += Res[(long)m * ldc + n];
          C[(long)m * ldc + n] = v;
        }
      }
}

// ---------------------------------------------------------------------------
// Fused multi-head self attention for one (b, head, 64-query tile).
// H: (B,L,DD) f32 shared QKV projection.
// Dynamic LDS layout (SW = L rounded up to 32):
//   S   : 64 x SW f32   score/prob tile
//   Sh  : 64 x SW f16   probabilities for P@V
//   sH  : SW x 32 f16   head slice (cols 16..31 zero)   -> A/B score frags
//   sHT : 16 x SW f16   transposed head slice           -> P@V B frags
// ---------------------------------------------------------------------------
__global__ __launch_bounds__(128)
void attention_kernel(const float* __restrict__ H, const float* __restrict__ mask,
                      float* __restrict__ O, int L, int SW, float scale) {
  extern __shared__ float smem[];
  float* S = smem;
  _Float16* Sh = (_Float16*)(S + 64 * SW);
  _Float16* sH = Sh + 64 * SW;
  _Float16* sHT = sH + (long)SW * 32;
  const int tid = threadIdx.x, lane = tid & 31, wv = tid >> 5;
  const int qt = blockIdx.x, head = blockIdx.y, b = blockIdx.z;
  const int q0 = qt * 64;
  const long base = (long)b * L * DD;
  const int l15 = lane & 15;
  const int kbA = (lane >= 16) ? 16 : 0, kbB = (lane >= 16) ? 16 : 0;
  const int mo = (lane >= 16) ? 8 : 0;

  // ---- stage head slice into LDS (zero-fill then bulk float4 convert) ----
  for (int idx = tid; idx < SW * 8; idx += 128) { v4h z = {}; *(v4h*)&sH[idx * 4] = z; }
  for (int idx = tid; idx < SW * 4; idx += 128) { v4h z = {}; *(v4h*)&sHT[idx * 4] = z; }
  __syncthreads();
  for (int idx = tid; idx < L * 4; idx += 128) {
    int key = idx >> 2, kg = (idx & 3) << 2;
    const float4 v = *(const float4*)&H[base + (long)key * DD + head * 16 + kg];
    v4h h;
    h[0] = (_Float16)v.x; h[1] = (_Float16)v.y;
    h[2] = (_Float16)v.z; h[3] = (_Float16)v.w;
    *(v4h*)&sH[key * 32 + kg] = h;
    sHT[(kg + 0) * SW + key] = h[0];
    sHT[(kg + 1) * SW + key] = h[1];
    sHT[(kg + 2) * SW + key] = h[2];
    sHT[(kg + 3) * SW + key] = h[3];
  }
  __syncthreads();

  // ---- scores: Q tile (16 rows per wave) x all keys ----------------------
  v16h aq;
  {
    int q = q0 + wv * 16 + l15;
    int qr = (q < SW) ? q : (SW - 1);  // clamped; masked rows never stored
    const v8h lo = *(const v8h*)&sH[qr * 32 + kbA];
    const v8h hi = *(const v8h*)&sH[qr * 32 + kbA + 8];  // zeros in lanes 16..31
#pragma unroll
    for (int t = 0; t < 8; ++t) { aq[t] = lo[t]; aq[t + 8] = hi[t]; }
  }
  const int KT = SW >> 4;
  for (int kt = 0; kt < KT; ++kt) {
    const v16h bk = *(const v16h*)&sH[(kt * 16 + l15) * 32 + kbB];
    v8f acc = {};
    acc = __builtin_amdgcn_wmma_f32_16x16x32_f16(false, aq, false, bk, (short)0,
                                                 acc, false, false);
#pragma unroll
    for (int r = 0; r < 8; ++r)
      S[(wv * 16 + r + mo) * SW + kt * 16 + l15] = acc[r];
  }
  __syncthreads();

  // ---- masked softmax per row (wave owns 16 rows) ------------------------
  for (int rr = 0; rr < 16; ++rr) {
    int q = q0 + wv * 16 + rr;
    float* row = &S[(wv * 16 + rr) * SW];
    float mq = (q < L) ? mask[b * L + q] : 0.f;
    float mx = -3.4e38f;
    for (int j = lane; j < SW; j += 32) {
      float mk = (j < L) ? mask[b * L + j] : 0.f;
      float s = (row[j] * scale * mk + (1.f - mk) * NEGV) * mq;
      if (j >= L) s = NEGV;
      row[j] = s;
      mx = fmaxf(mx, s);
    }
    for (int o = 16; o; o >>= 1) mx = fmaxf(mx, __shfl_xor(mx, o, 32));
    float sum = 0.f;
    for (int j = lane; j < SW; j += 32) {
      float e = __expf(row[j] - mx);
      row[j] = e;
      sum += e;
    }
    for (int o = 16; o; o >>= 1) sum += __shfl_xor(sum, o, 32);
    float inv = 1.f / sum;
    _Float16* hrow = &Sh[(wv * 16 + rr) * SW];
    for (int j = lane; j < SW; j += 32) hrow[j] = (_Float16)(row[j] * inv);
  }
  __syncthreads();

  // ---- O = P @ V : each wave -> 16 q rows x 16 d cols, K = SW ------------
  v8f o = {};
  for (int kc = 0; kc < SW; kc += 32) {
    v16h a;
    {
      const v8h lo = *(const v8h*)&Sh[(wv * 16 + l15) * SW + kc + kbA];
      const v8h hi = *(const v8h*)&Sh[(wv * 16 + l15) * SW + kc + kbA + 8];
#pragma unroll
      for (int t = 0; t < 8; ++t) { a[t] = lo[t]; a[t + 8] = hi[t]; }
    }
    const v16h bv = *(const v16h*)&sHT[l15 * SW + kc + kbB];
    o = __builtin_amdgcn_wmma_f32_16x16x32_f16(false, a, false, bv, (short)0, o,
                                               false, false);
  }
#pragma unroll
  for (int r = 0; r < 8; ++r) {
    int q = q0 + wv * 16 + r + mo;
    if (q < L) O[base + (long)q * DD + head * 16 + l15] = o[r];
  }
}

// ---------------------------------------------------------------------------
// CQ attention trilinear scores via WMMA: S[b,i,j] (j padded to 64).
// C*wcq and Q tiles are staged in LDS so fragments are vector ds loads.
// ---------------------------------------------------------------------------
__global__ __launch_bounds__(128)
void cq_score_kernel(const float* __restrict__ Cx, const float* __restrict__ Qx,
                     const float* __restrict__ wcq, const float* __restrict__ cwc,
                     const float* __restrict__ qwq, float* __restrict__ S) {
  __shared__ _Float16 sCw[64 * 128];
  __shared__ _Float16 sQ[64 * 128];
  const int tid = threadIdx.x, lane = tid & 31, wv = tid >> 5, l15 = lane & 15;
  const int it = blockIdx.x, b = blockIdx.z;
  const int kbA = (lane >= 16) ? 16 : 0, kbB = (lane >= 16) ? 16 : 0;
  const int mo = (lane >= 16) ? 8 : 0;
  const int i0 = it * 64;

  for (int g = tid; g < 64 * 32; g += 128) {
    int r = g >> 5, kg = (g & 31) << 2;
    // Q tile (rows j, zero-padded past LQ)
    v4h hq = {};
    if (r < LQ) {
      const float4 v = *(const float4*)&Qx[((long)b * LQ + r) * DD + kg];
      hq[0] = (_Float16)v.x; hq[1] = (_Float16)v.y;
      hq[2] = (_Float16)v.z; hq[3] = (_Float16)v.w;
    }
    *(v4h*)&sQ[r * 128 + kg] = hq;
    // C tile scaled by wcq (rows i, zero-padded past LC)
    v4h hc = {};
    int i = i0 + r;
    if (i < LC) {
      const float4 v = *(const float4*)&Cx[((long)b * LC + i) * DD + kg];
      const float4 w = *(const float4*)&wcq[kg];
      hc[0] = (_Float16)(v.x * w.x); hc[1] = (_Float16)(v.y * w.y);
      hc[2] = (_Float16)(v.z * w.z); hc[3] = (_Float16)(v.w * w.w);
    }
    *(v4h*)&sCw[r * 128 + kg] = hc;
  }
  __syncthreads();

  for (int jt = 0; jt < 4; ++jt) {
    v8f acc = {};
    for (int kc = 0; kc < 128; kc += 32) {
      v16h a;
      {
        const v8h lo = *(const v8h*)&sCw[(wv * 16 + l15) * 128 + kc + kbA];
        const v8h hi = *(const v8h*)&sCw[(wv * 16 + l15) * 128 + kc + kbA + 8];
#pragma unroll
        for (int t = 0; t < 8; ++t) { a[t] = lo[t]; a[t + 8] = hi[t]; }
      }
      const v16h bq = *(const v16h*)&sQ[(jt * 16 + l15) * 128 + kc + kbB];
      acc = __builtin_amdgcn_wmma_f32_16x16x32_f16(false, a, false, bq, (short)0,
                                                   acc, false, false);
    }
#pragma unroll
    for (int r = 0; r < 8; ++r) {
      int i = i0 + wv * 16 + r + mo, j = jt * 16 + l15;
      if (i < LC)
        S[((long)b * LC + i) * 64 + j] =
            acc[r] + cwc[(long)b * LC + i] + ((j < LQ) ? qwq[(long)b * LQ + j] : 0.f);
    }
  }
}

// ---------------------------------------------------------------------------
// Elementwise / reduction helper kernels
// ---------------------------------------------------------------------------
__global__ void wT_to_f16_kernel(const float* __restrict__ W, _Float16* __restrict__ B,
                                 int O, int I, int KP, int NP) {
  long idx = (long)blockIdx.x * blockDim.x + threadIdx.x;
  if (idx >= (long)KP * NP) return;
  int n = idx % NP;
  int k = idx / NP;
  B[idx] = (k < I && n < O) ? (_Float16)W[(long)n * I + k] : (_Float16)0.f;
}

__global__ void pack_rows_f16_kernel(const float* __restrict__ X,
                                     _Float16* __restrict__ Y, int RS, int R,
                                     int RP, int Ccols) {
  long idx = (long)blockIdx.x * blockDim.x + threadIdx.x;
  long tot = (long)BBATCH * RP * Ccols;
  if (idx >= tot) return;
  int c = idx % Ccols;
  long p = idx / Ccols;
  int r = p % RP;
  int b = p / RP;
  Y[idx] = (r < R) ? (_Float16)X[((long)b * RS + r) * Ccols + c] : (_Float16)0.f;
}

__global__ void mask_kernel(const int* __restrict__ wid, float* __restrict__ m, long n) {
  long idx = (long)blockIdx.x * blockDim.x + threadIdx.x;
  if (idx < n) m[idx] = (wid[idx] != 0) ? 1.f : 0.f;
}

__global__ void gather_char_kernel(const int* __restrict__ cid,
                                   const float* __restrict__ char_mat,
                                   float* __restrict__ out, int L) {
  long idx = (long)blockIdx.x * blockDim.x + threadIdx.x;
  if (idx >= (long)BBATCH * L * 16 * 64) return;
  int c = idx & 63;
  long p = idx >> 6;
  int w = p & 15;
  long bl = p >> 4;
  out[idx] = char_mat[(long)cid[bl * 16 + w] * 64 + c];
}

__global__ void dwconv2d_kernel(const float* __restrict__ X, float* __restrict__ Y,
                                const float* __restrict__ Wt,
                                const float* __restrict__ bias, int L) {
  long idx = (long)blockIdx.x * blockDim.x + threadIdx.x;
  if (idx >= (long)BBATCH * L * 16 * 64) return;
  int c = idx & 63;
  long p = idx >> 6;
  int w = p & 15;
  long bl = p >> 4;
  int l = bl % L;
  long b = bl / L;
  float acc = bias[c];
  for (int dh = 0; dh < 5; ++dh) {
    int ll = l + dh - 2;
    if (ll < 0 || ll >= L) continue;
    for (int dw = 0; dw < 5; ++dw) {
      int ww = w + dw - 2;
      if (ww < 0 || ww >= 16) continue;
      acc += X[(((long)b * L + ll) * 16 + ww) * 64 + c] * Wt[c * 25 + dh * 5 + dw];
    }
  }
  Y[idx] = acc;
}

__global__ void charmax_kernel(const float* __restrict__ X, float* __restrict__ Y, int L) {
  long idx = (long)blockIdx.x * blockDim.x + threadIdx.x;
  if (idx >= (long)BBATCH * L * 64) return;
  int c = idx & 63;
  long bl = idx >> 6;
  float m = -3.4e38f;
  for (int w = 0; w < 16; ++w) m = fmaxf(m, X[(bl * 16 + w) * 64 + c]);
  Y[idx] = m;
}

__global__ void concat_emb_kernel(const float* __restrict__ chemb,
                                  const int* __restrict__ wid,
                                  const float* __restrict__ word_mat,
                                  float* __restrict__ out, int L) {
  long idx = (long)blockIdx.x * blockDim.x + threadIdx.x;
  if (idx >= (long)BBATCH * L * EP) return;
  int c = idx % EP;
  long bl = idx / EP;
  float v = 0.f;
  if (c < 64) v = chemb[bl * 64 + c];
  else if (c < EE) v = word_mat[(long)wid[bl] * 300 + (c - 64)];
  out[idx] = v;
}

__global__ void hw_combine_kernel(float* __restrict__ X, const float* __restrict__ G,
                                  const float* __restrict__ Nl, long n) {
  long idx = (long)blockIdx.x * blockDim.x + threadIdx.x;
  if (idx >= n) return;
  float g = G[idx];
  X[idx] = g * Nl[idx] + (1.f - g) * X[idx];
}

__global__ void dwconv1d_kernel(const float* __restrict__ X, float* __restrict__ Y,
                                const float* __restrict__ Wt,
                                const float* __restrict__ bias, int L, int Cp,
                                int Cr, int k) {
  long idx = (long)blockIdx.x * blockDim.x + threadIdx.x;
  if (idx >= (long)BBATCH * L * Cp) return;
  int c = idx % Cp;
  long bl = idx / Cp;
  int l = bl % L;
  long b = bl / L;
  if (c >= Cr) { Y[idx] = 0.f; return; }
  float acc = bias[c];
  int h = k >> 1;
  for (int t = 0; t < k; ++t) {
    int ll = l + t - h;
    if (ll >= 0 && ll < L) acc += X[((long)b * L + ll) * Cp + c] * Wt[c * k + t];
  }
  Y[idx] = acc;
}

__global__ void posadd_kernel(float* __restrict__ X, const float* __restrict__ pos, int L) {
  long idx = (long)blockIdx.x * blockDim.x + threadIdx.x;
  if (idx >= (long)BBATCH * L * DD) return;
  int d = idx & 127;
  long bl = idx >> 7;
  int l = bl % L;
  X[idx] += pos[l * DD + d];
}

__global__ void layernorm_kernel(float* __restrict__ X, const float* __restrict__ g,
                                 const float* __restrict__ bta, int L) {
  const int b = blockIdx.x, tid = threadIdx.x;
  const long base = (long)b * L * DD;
  const int n = L * DD;
  __shared__ float sh1[256], sh2[256];
  float s = 0.f, s2 = 0.f;
  for (int i = tid; i < n; i += 256) {
    float v = X[base + i];
    s += v;
    s2 += v * v;
  }
  sh1[tid] = s;
  sh2[tid] = s2;
  __syncthreads();
  for (int o = 128; o; o >>= 1) {
    if (tid < o) { sh1[tid] += sh1[tid + o]; sh2[tid] += sh2[tid + o]; }
    __syncthreads();
  }
  float mu = sh1[0] / n;
  float var = sh2[0] / n - mu * mu;
  float rstd = rsqrtf(var + 1e-5f);
  for (int i = tid; i < n; i += 256) {
    int l = i / DD, d = i % DD;
    X[base + i] = (X[base + i] - mu) * rstd * g[(long)d * L + l] + bta[(long)d * L + l];
  }
}

__global__ void rowdot_kernel(const float* __restrict__ X, const float* __restrict__ w,
                              float* __restrict__ out, long rows) {
  long row = (long)blockIdx.x * 4 + (threadIdx.x >> 5);
  if (row >= rows) return;
  int lane = threadIdx.x & 31;
  float s = 0.f;
  for (int t = 0; t < 4; ++t) s += X[row * DD + lane + 32 * t] * w[lane + 32 * t];
  for (int o = 16; o; o >>= 1) s += __shfl_xor(s, o, 32);
  if (lane == 0) out[row] = s;
}

// softmax over j (Lq) per context row; output padded to 64
__global__ void cq_softmax_row_kernel(const float* __restrict__ S,
                                      const float* __restrict__ qmask,
                                      float* __restrict__ S1) {
  long row = (long)blockIdx.x * 4 + (threadIdx.x >> 5);
  if (row >= (long)BBATCH * LC) return;
  int b = row / LC;
  int lane = threadIdx.x & 31;
  float v[2];
  float mx = -3.4e38f;
  for (int t = 0; t < 2; ++t) {
    int j = lane + 32 * t;
    float s = NEGV;
    if (j < LQ) {
      float mk = qmask[b * LQ + j];
      s = S[row * 64 + j] * mk + (1.f - mk) * NEGV;
    }
    v[t] = s;
    mx = fmaxf(mx, s);
  }
  for (int o = 16; o; o >>= 1) mx = fmaxf(mx, __shfl_xor(mx, o, 32));
  float sum = 0.f;
  for (int t = 0; t < 2; ++t) { v[t] = __expf(v[t] - mx); sum += v[t]; }
  for (int o = 16; o; o >>= 1) sum += __shfl_xor(sum, o, 32);
  float inv = 1.f / sum;
  for (int t = 0; t < 2; ++t) {
    int j = lane + 32 * t;
    S1[row * 64 + j] = (j < LQ) ? v[t] * inv : 0.f;
  }
}

// softmax over i (Lc) per (b,j); writes transposed+padded S2t (B,64,416)
__global__ void cq_softmax_col_kernel(const float* __restrict__ S,
                                      const float* __restrict__ cmask,
                                      float* __restrict__ S2t) {
  long idx = (long)blockIdx.x * 4 + (threadIdx.x >> 5);
  if (idx >= (long)BBATCH * 64) return;
  int b = idx >> 6, j = idx & 63;
  int lane = threadIdx.x & 31;
  float v[13];
  float mx = -3.4e38f;
  for (int t = 0; t < 13; ++t) {
    int i = lane + 32 * t;
    float s = NEGV;
    if (i < LC && j < LQ) {
      float mk = cmask[b * LC + i];
      s = S[((long)b * LC + i) * 64 + j] * mk + (1.f - mk) * NEGV;
    }
    v[t] = s;
    mx = fmaxf(mx, s);
  }
  for (int o = 16; o; o >>= 1) mx = fmaxf(mx, __shfl_xor(mx, o, 32));
  float sum = 0.f;
  for (int t = 0; t < 13; ++t) { v[t] = __expf(v[t] - mx); sum += v[t]; }
  for (int o = 16; o; o >>= 1) sum += __shfl_xor(sum, o, 32);
  float inv = 1.f / sum;
  for (int t = 0; t < 13; ++t) {
    int i = lane + 32 * t;
    if (i < 416)
      S2t[((long)b * 64 + j) * 416 + i] = (i < LC && j < LQ) ? v[t] * inv : 0.f;
  }
}

__global__ void concat4_kernel(const float* __restrict__ Cx, const float* __restrict__ Ax,
                               const float* __restrict__ Bx, float* __restrict__ X) {
  long idx = (long)blockIdx.x * blockDim.x + threadIdx.x;
  if (idx >= (long)BBATCH * LC * 512) return;
  int c = idx & 511;
  long bl = idx >> 9;
  float v;
  if (c < 128) v = Cx[bl * 128 + c];
  else if (c < 256) v = Ax[bl * 128 + (c - 128)];
  else if (c < 384) v = Cx[bl * 128 + (c - 256)] * Ax[bl * 128 + (c - 256)];
  else v = Cx[bl * 128 + (c - 384)] * Bx[bl * 128 + (c - 384)];
  X[idx] = v;
}

// masked logits + log_softmax over length (one block per batch sample)
__global__ void logits_kernel(const float* __restrict__ Ma, const float* __restrict__ Mb,
                              const float* __restrict__ w, const float* __restrict__ mask,
                              float* __restrict__ out, int L) {
  const int b = blockIdx.x, tid = threadIdx.x;
  __shared__ float sh[512];
  float v = -3.4e38f;
  if (tid < L) {
    const float* m1 = &Ma[((long)b * L + tid) * DD];
    const float* m2 = &Mb[((long)b * L + tid) * DD];
    float s = 0.f;
    for (int d = 0; d < DD; ++d) s += w[d] * m1[d] + w[DD + d] * m2[d];
    float mk = mask[b * L + tid];
    v = s * mk + (1.f - mk) * NEGV;
  }
  sh[tid] = v;
  __syncthreads();
  for (int o = 256; o; o >>= 1) {
    if (tid < o) sh[tid] = fmaxf(sh[tid], sh[tid + o]);
    __syncthreads();
  }
  float mx = sh[0];
  __syncthreads();
  float e = (tid < L) ? __expf(v - mx) : 0.f;
  sh[tid] = e;
  __syncthreads();
  for (int o = 256; o; o >>= 1) {
    if (tid < o) sh[tid] += sh[tid + o];
    __syncthreads();
  }
  float lse = logf(sh[0]);
  if (tid < L) out[(long)b * L + tid] = (v - mx) - lse;
}

// ---------------------------------------------------------------------------
// Host side
// ---------------------------------------------------------------------------
static void launch_gemm(hipStream_t st, const float* A, long sA, int lda,
                        const _Float16* Bm, long sB, float* C, long sC, int ldc,
                        const float* bias, int biasN, const float* Res, long sR,
                        int M, int N, int K, int KP, int NP, int act, int batch) {
  dim3 g((N + 63) / 64, (M + 63) / 64, batch);
  gemm_f16_kernel<<<g, 128, 0, st>>>(A, sA, lda, Bm, sB, C, sC, ldc, bias, biasN,
                                     Res, sR, M, N, K, KP, NP, act);
}

struct EncP {
  const float *pos, *fc_b, *normb_g, *normb_b, *norms_g, *norms_b, *norme_g, *norme_b;
  const float *dw_w[4], *dw_b[4], *pw_b[4];
  const _Float16 *pwT[4], *WqsT, *WoT, *fcT;
  int nconv, k;
};

// Assumed depth-first flattening of the setup_inputs() dict (102 leaf tensors)
enum {
  IN_CWID = 0, IN_CCID, IN_QWID, IN_QCID,
  IN_WORD_MAT, IN_CHAR_MAT,
  IN_E2D_DW_W, IN_E2D_DW_B, IN_E2D_PW_W, IN_E2D_PW_B,
  IN_HW_LIN_W, IN_HW_LIN_B, IN_HW_GATE_W, IN_HW_GATE_B,
  IN_CC_DW_W, IN_CC_DW_B, IN_CC_PW_W, IN_CC_PW_B,
  IN_QC_DW_W, IN_QC_DW_B, IN_QC_PW_W, IN_QC_PW_B,
  IN_CENC = 22,       // 27 tensors
  IN_QENC = 49,       // 27 tensors
  IN_CQ_W = 76,
  IN_RS_DW_W = 77, IN_RS_DW_B, IN_RS_PW_W, IN_RS_PW_B,
  IN_MENC = 81,       // 19 tensors
  IN_W1 = 100, IN_W2 = 101
};

extern "C" void kernel_launch(void* const* d_in, const int* in_sizes, int n_in,
                              void* d_out, int out_size, void* d_ws, size_t ws_size,
                              hipStream_t stream) {
  (void)in_sizes; (void)out_size; (void)ws_size;
  const float* F[102];
  for (int i = 0; i < 102 && i < n_in; ++i) F[i] = (const float*)d_in[i];
  const int* Cwid = (const int*)d_in[IN_CWID];
  const int* Ccid = (const int*)d_in[IN_CCID];
  const int* Qwid = (const int*)d_in[IN_QWID];
  const int* Qcid = (const int*)d_in[IN_QCID];

  size_t off = 0;
  auto alloc = [&](size_t bytes) -> void* {
    void* p = (char*)d_ws + off;
    off = (off + bytes + 255) & ~(size_t)255;
    return p;
  };
  auto convW = [&](const float* W, int O, int I, int KP, int NP) -> _Float16* {
    _Float16* p = (_Float16*)alloc((size_t)KP * NP * 2);
    wT_to_f16_kernel<<<g1d((long)KP * NP), 256, 0, stream>>>(W, p, O, I, KP, NP);
    return p;
  };

  // ---- convert weights to padded f16 B-matrices --------------------------
  _Float16* emb_pwT = convW(F[IN_E2D_PW_W], 64, 64, 64, 64);
  _Float16* hw_gateT[2], *hw_linT[2];
  for (int i = 0; i < 2; ++i) {
    hw_gateT[i] = convW(F[IN_HW_GATE_W] + (long)i * EE * EE, EE, EE, EP, EP);
    hw_linT[i]  = convW(F[IN_HW_LIN_W]  + (long)i * EE * EE, EE, EE, EP, EP);
  }
  _Float16* cc_pwT = convW(F[IN_CC_PW_W], DD, EE, EP, DD);
  _Float16* qc_pwT = convW(F[IN_QC_PW_W], DD, EE, EP, DD);
  _Float16* rs_pwT = convW(F[IN_RS_PW_W], DD, 512, 512, DD);

  auto makeEnc = [&](int base, int nconv, int k) -> EncP {
    EncP e{};
    e.nconv = nconv; e.k = k;
    int idx = base;
    e.pos = F[idx++];
    for (int i = 0; i < nconv; ++i) {
      e.dw_w[i] = F[idx];
      e.dw_b[i] = F[idx + 1];
      e.pwT[i] = convW(F[idx + 2], DD, DD, DD, DD);
      e.pw_b[i] = F[idx + 3];
      idx += 4;
    }
    e.WqsT = convW(F[idx++], DD, DD, DD, DD);
    e.WoT  = convW(F[idx++], DD, DD, DD, DD);
    e.fcT  = convW(F[idx++], DD, DD, DD, DD);
    e.fc_b = F[idx++];
    e.normb_g = F[idx++]; e.normb_b = F[idx++];
    e.norms_g = F[idx++]; e.norms_b = F[idx++];
    e.norme_g = F[idx++]; e.norme_b = F[idx++];
    return e;
  };
  EncP cencP = makeEnc(IN_CENC, 4, 7);
  EncP qencP = makeEnc(IN_QENC, 4, 7);
  EncP mencP = makeEnc(IN_MENC, 2, 5);

  // ---- activation workspace ----------------------------------------------
  float* cmask = (float*)alloc((size_t)BBATCH * LC * 4);
  float* qmask = (float*)alloc((size_t)BBATCH * LQ * 4);
  float* chbuf = (float*)alloc((size_t)BBATCH * LC * 16 * 64 * 4);
  float* dwch  = (float*)alloc((size_t)BBATCH * LC * 16 * 64 * 4);
  float* chemb = (float*)alloc((size_t)BBATCH * LC * 64 * 4);
  float* embC  = (float*)alloc((size_t)BBATCH * LC * EP * 4);
  float* embQ  = (float*)alloc((size_t)BBATCH * LQ * EP * 4);
  float* gbuf  = (float*)alloc((size_t)BBATCH * LC * EP * 4);
  float* nbuf  = (float*)alloc((size_t)BBATCH * LC * EP * 4);
  float* Cenc  = (float*)alloc((size_t)BBATCH * LC * DD * 4);
  float* Qenc  = (float*)alloc((size_t)BBATCH * LQ * DD * 4);
  float* tb = (float*)alloc((size_t)BBATCH * LC * DD * 4);
  float* ub = (float*)alloc((size_t)BBATCH * LC * DD * 4);
  float* rb = (float*)alloc((size_t)BBATCH * LC * DD * 4);
  float* hb = (float*)alloc((size_t)BBATCH * LC * DD * 4);
  float* ab = (float*)alloc((size_t)BBATCH * LC * DD * 4);
  float* cwcB = (float*)alloc((size_t)BBATCH * LC * 4);
  float* qwqB = (float*)alloc((size_t)BBATCH * LQ * 4);
  float* Sbuf  = (float*)alloc((size_t)BBATCH * LC * 64 * 4);
  float* S1buf = (float*)alloc((size_t)BBATCH * LC * 64 * 4);
  float* S2t   = (float*)alloc((size_t)BBATCH * 64 * 416 * 4);
  float* Tbuf  = (float*)alloc((size_t)BBATCH * 64 * DD * 4);
  _Float16* Qh = (_Float16*)alloc((size_t)BBATCH * 64 * DD * 2);
  _Float16* Ch = (_Float16*)alloc((size_t)BBATCH * 416 * DD * 2);
  _Float16* Th = (_Float16*)alloc((size_t)BBATCH * 64 * DD * 2);
  float* Abuf  = (float*)alloc((size_t)BBATCH * LC * DD * 4);
  float* Bmbuf = (float*)alloc((size_t)BBATCH * LC * DD * 4);
  float* Xbuf  = (float*)alloc((size_t)BBATCH * LC * 512 * 4);
  float* Xdw   = (float*)alloc((size_t)BBATCH * LC * 512 * 4);
  float* M1s   = (float*)alloc((size_t)BBATCH * LC * DD * 4);
  float* M2s   = (float*)alloc((size_t)BBATCH * LC * DD * 4);
  float* Wb    = (float*)alloc((size_t)BBATCH * LC * DD * 4);

  mask_kernel<<<g1d((long)BBATCH * LC), 256, 0, stream>>>(Cwid, cmask, (long)BBATCH * LC);
  mask_kernel<<<g1d((long)BBATCH * LQ), 256, 0, stream>>>(Qwid, qmask, (long)BBATCH * LQ);

  // ---- embedding: char conv + word gather + highway + input conv ---------
  auto embed_side = [&](const int* wid, const int* cid, int L, float* embX,
                        const _Float16* pwT_in, const float* dw_w,
                        const float* dw_b, const float* pw_b, float* encOut) {
    long tot = (long)BBATCH * L * 16 * 64;
    gather_char_kernel<<<g1d(tot), 256, 0, stream>>>(cid, F[IN_CHAR_MAT], chbuf, L);
    dwconv2d_kernel<<<g1d(tot), 256, 0, stream>>>(chbuf, dwch, F[IN_E2D_DW_W],
                                                  F[IN_E2D_DW_B], L);
    launch_gemm(stream, dwch, 0, 64, emb_pwT, 0, chbuf, 0, 64, F[IN_E2D_PW_B], 64,
                nullptr, 0, BBATCH * L * 16, 64, 64, 64, 64, 1, 1);
    charmax_kernel<<<g1d((long)BBATCH * L * 64), 256, 0, stream>>>(chbuf, chemb, L);
    concat_emb_kernel<<<g1d((long)BBATCH * L * EP), 256, 0, stream>>>(
        chemb, wid, F[IN_WORD_MAT], embX, L);
    for (int i = 0; i < 2; ++i) {
      launch_gemm(stream, embX, 0, EP, hw_gateT[i], 0, gbuf, 0, EP,
                  F[IN_HW_GATE_B] + (long)i * EE, EE, nullptr, 0, BBATCH * L, EP,
                  EE, EP, EP, 2, 1);
      launch_gemm(stream, embX, 0, EP, hw_linT[i], 0, nbuf, 0, EP,
                  F[IN_HW_LIN_B] + (long)i * EE, EE, nullptr, 0, BBATCH * L, EP,
                  EE, EP, EP, 1, 1);
      hw_combine_kernel<<<g1d((long)BBATCH * L * EP), 256, 0, stream>>>(
          embX, gbuf, nbuf, (long)BBATCH * L * EP);
    }
    dwconv1d_kernel<<<g1d((long)BBATCH * L * EP), 256, 0, stream>>>(
        embX, gbuf, dw_w, dw_b, L, EP, EE, 5);
    launch_gemm(stream, gbuf, 0, EP, pwT_in, 0, encOut, 0, DD, pw_b, DD, nullptr,
                0, BBATCH * L, DD, EE, EP, DD, 0, 1);
  };
  embed_side(Cwid, Ccid, LC, embC, cc_pwT, F[IN_CC_DW_W], F[IN_CC_DW_B], F[IN_CC_PW_B], Cenc);
  embed_side(Qwid, Qcid, LQ, embQ, qc_pwT, F[IN_QC_DW_W], F[IN_QC_DW_B], F[IN_QC_PW_B], Qenc);

  // ---- encoder block ------------------------------------------------------
  auto run_encoder = [&](float* X, const float* mask, const EncP& e, int L) {
    long rows = (long)BBATCH * L;
    size_t bytes = rows * DD * sizeof(float);
    int SW = ((L + 31) / 32) * 32;
    posadd_kernel<<<g1d(rows * DD), 256, 0, stream>>>(X, e.pos, L);
    hipMemcpyAsync(rb, X, bytes, hipMemcpyDeviceToDevice, stream);
    layernorm_kernel<<<BBATCH, 256, 0, stream>>>(X, e.normb_g, e.normb_b, L);
    for (int i = 0; i < e.nconv; ++i) {
      dwconv1d_kernel<<<g1d(rows * DD), 256, 0, stream>>>(X, tb, e.dw_w[i],
                                                          e.dw_b[i], L, DD, DD, e.k);
      launch_gemm(stream, tb, 0, DD, e.pwT[i], 0, X, 0, DD, e.pw_b[i], DD, rb, 0,
                  (int)rows, DD, DD, DD, DD, 1, 1);
      hipMemcpyAsync(rb, X, bytes, hipMemcpyDeviceToDevice, stream);
      layernorm_kernel<<<BBATCH, 256, 0, stream>>>(X, e.norms_g + (long)i * DD * L,
                                                   e.norms_b + (long)i * DD * L, L);
    }
    launch_gemm(stream, X, 0, DD, e.WqsT, 0, hb, 0, DD, nullptr, 0, nullptr, 0,
                (int)rows, DD, DD, DD, DD, 0, 1);
    attention_kernel<<<dim3((L + 63) / 64, NH, BBATCH), 128, (size_t)SW * 480,
                       stream>>>(hb, mask, ab, L, SW, 0.25f);
    launch_gemm(stream, ab, 0, DD, e.WoT, 0, X, 0, DD, nullptr, 0, rb, 0,
                (int)rows, DD, DD, DD, DD, 0, 1);
    hipMemcpyAsync(rb, X, bytes, hipMemcpyDeviceToDevice, stream);
    layernorm_kernel<<<BBATCH, 256, 0, stream>>>(X, e.norme_g, e.norme_b, L);
    launch_gemm(stream, X, 0, DD, e.fcT, 0, ub, 0, DD, e.fc_b, DD, rb, 0,
                (int)rows, DD, DD, DD, DD, 1, 1);
    hipMemcpyAsync(X, ub, bytes, hipMemcpyDeviceToDevice, stream);
  };

  run_encoder(Cenc, cmask, cencP, LC);
  run_encoder(Qenc, qmask, qencP, LQ);

  // ---- context-query attention -------------------------------------------
  rowdot_kernel<<<dim3((BBATCH * LC + 3) / 4), 128, 0, stream>>>(
      Cenc, F[IN_CQ_W], cwcB, (long)BBATCH * LC);
  rowdot_kernel<<<dim3((BBATCH * LQ + 3) / 4), 128, 0, stream>>>(
      Qenc, F[IN_CQ_W] + DD, qwqB, (long)BBATCH * LQ);
  cq_score_kernel<<<dim3((LC + 63) / 64, 1, BBATCH), 128, 0, stream>>>(
      Cenc, Qenc, F[IN_CQ_W] + 2 * DD, cwcB, qwqB, Sbuf);
  cq_softmax_row_kernel<<<dim3((BBATCH * LC + 3) / 4), 128, 0, stream>>>(Sbuf, qmask, S1buf);
  cq_softmax_col_kernel<<<dim3((BBATCH * 64 + 3) / 4), 128, 0, stream>>>(Sbuf, cmask, S2t);
  pack_rows_f16_kernel<<<g1d((long)BBATCH * 64 * DD), 256, 0, stream>>>(Qenc, Qh, LQ, LQ, 64, DD);
  pack_rows_f16_kernel<<<g1d((long)BBATCH * 416 * DD), 256, 0, stream>>>(Cenc, Ch, LC, LC, 416, DD);
  // A = S1 @ Q
  launch_gemm(stream, S1buf, (long)LC * 64, 64, Qh, (long)64 * DD, Abuf,
              (long)LC * DD, DD, nullptr, 0, nullptr, 0, LC, DD, LQ, 64, DD, 0, BBATCH);
  // T = S2^T @ C  (re-associated: B = S1 @ T instead of S1 @ S2^T @ C)
  launch_gemm(stream, S2t, (long)64 * 416, 416, Ch, (long)416 * DD, Tbuf,
              (long)64 * DD, DD, nullptr, 0, nullptr, 0, LQ, DD, LC, 416, DD, 0, BBATCH);
  pack_rows_f16_kernel<<<g1d((long)BBATCH * 64 * DD), 256, 0, stream>>>(Tbuf, Th, 64, LQ, 64, DD);
  launch_gemm(stream, S1buf, (long)LC * 64, 64, Th, (long)64 * DD, Bmbuf,
              (long)LC * DD, DD, nullptr, 0, nullptr, 0, LC, DD, LQ, 64, DD, 0, BBATCH);
  concat4_kernel<<<g1d((long)BBATCH * LC * 512), 256, 0, stream>>>(Cenc, Abuf, Bmbuf, Xbuf);

  // ---- cq resizer + 3 x 7 shared model encoder stacks ---------------------
  dwconv1d_kernel<<<g1d((long)BBATCH * LC * 512), 256, 0, stream>>>(
      Xbuf, Xdw, F[IN_RS_DW_W], F[IN_RS_DW_B], LC, 512, 512, 5);
  launch_gemm(stream, Xdw, 0, 512, rs_pwT, 0, Wb, 0, DD, F[IN_RS_PW_B], DD,
              nullptr, 0, BBATCH * LC, DD, 512, 512, DD, 0, 1);

  size_t mbytes = (size_t)BBATCH * LC * DD * sizeof(float);
  for (int i = 0; i < 7; ++i) run_encoder(Wb, cmask, mencP, LC);
  hipMemcpyAsync(M1s, Wb, mbytes, hipMemcpyDeviceToDevice, stream);
  for (int i = 0; i < 7; ++i) run_encoder(Wb, cmask, mencP, LC);
  hipMemcpyAsync(M2s, Wb, mbytes, hipMemcpyDeviceToDevice, stream);
  for (int i = 0; i < 7; ++i) run_encoder(Wb, cmask, mencP, LC);  // Wb = M3

  // ---- output heads: masked logits + log_softmax --------------------------
  float* out = (float*)d_out;
  logits_kernel<<<BBATCH, 512, 0, stream>>>(M1s, M2s, F[IN_W1], cmask, out, LC);
  logits_kernel<<<BBATCH, 512, 0, stream>>>(M1s, Wb, F[IN_W2], cmask,
                                            out + (long)BBATCH * LC, LC);
}
